// CrossDomainAspectLabelPropagation_17781164605538
// MI455X (gfx1250) — compile-verified, matrix operand order checked by reading, not run on hardware
//
#include <hip/hip_runtime.h>

#define HH 768
#define BB 8192
#define DD 5
#define BM 128
#define MT  (BB / BM + DD)     // 69 worst-case padded row-tiles
#define MPAD (MT * BM)         // 8832

typedef __attribute__((ext_vector_type(16))) __bf16 v16bf;
typedef __attribute__((ext_vector_type(8)))  __bf16 v8bf;
typedef __attribute__((ext_vector_type(8)))  float  v8f;

__device__ __forceinline__ unsigned short f2bf(float f) {
    unsigned u = __float_as_uint(f);
    u += 0x7fffu + ((u >> 16) & 1u);            // round-to-nearest-even
    return (unsigned short)(u >> 16);
}
__device__ __forceinline__ float bf2f_lo(unsigned u) { return __uint_as_float(u << 16); }
__device__ __forceinline__ float bf2f_hi(unsigned u) { return __uint_as_float(u & 0xffff0000u); }

// ---------------------------------------------------------------- fp32 -> bf16
__global__ void k_cvt(const float* __restrict__ in, unsigned short* __restrict__ out, int n) {
    int i = blockIdx.x * blockDim.x + threadIdx.x;
    int stride = gridDim.x * blockDim.x;
    for (; i < n; i += stride) out[i] = f2bf(in[i]);
}

// ------------------- tiled transpose + convert: WT[n*K+k] = bf16(W[k*N+n])
__global__ __launch_bounds__(256)
void k_cvtT(const float* __restrict__ W, unsigned short* __restrict__ WT, int K, int N) {
    __shared__ float tile[32][33];
    const float* Wb = W + (size_t)blockIdx.z * K * N;
    unsigned short* WTb = WT + (size_t)blockIdx.z * K * N;
    int n0 = blockIdx.x * 32, k0 = blockIdx.y * 32;
    int tx = threadIdx.x, ty = threadIdx.y;     // 32 x 8
#pragma unroll
    for (int j = 0; j < 32; j += 8)
        tile[ty + j][tx] = Wb[(size_t)(k0 + ty + j) * N + n0 + tx];
    __syncthreads();
#pragma unroll
    for (int j = 0; j < 32; j += 8)
        WTb[(size_t)(n0 + ty + j) * K + k0 + tx] = f2bf(tile[tx][ty + j]);
}

// --------------------------------------- counting sort of rows by domain (1 WG)
__global__ void k_sort(const int* __restrict__ dom, int* __restrict__ perm,
                       int* __restrict__ tileDom, float* __restrict__ lossAcc,
                       unsigned short* __restrict__ zrow) {
    __shared__ int cnt[DD];
    __shared__ int starts[DD + 1];
    __shared__ int pos[DD];
    int t = threadIdx.x;
    if (t < DD) cnt[t] = 0;
    if (t == 0) lossAcc[0] = 0.f;               // zero loss accumulator every call
    for (int i = t; i < 2 * HH; i += blockDim.x) zrow[i] = 0;  // zero gather row
    __syncthreads();
    for (int i = t; i < BB; i += blockDim.x) atomicAdd(&cnt[dom[i]], 1);
    __syncthreads();
    if (t == 0) {
        int s = 0;
        for (int d = 0; d < DD; ++d) { starts[d] = s; s += (cnt[d] + BM - 1) & ~(BM - 1); }
        starts[DD] = s;
    }
    __syncthreads();
    for (int i = t; i < MPAD; i += blockDim.x) perm[i] = -1;
    for (int i = t; i < MT; i += blockDim.x) {
        int d0 = 0;
        for (int d = 0; d < DD; ++d)
            if (i * BM >= starts[d] && i * BM < starts[d + 1]) d0 = d;
        tileDom[i] = d0;
    }
    if (t < DD) pos[t] = starts[t];
    __syncthreads();
    for (int i = t; i < BB; i += blockDim.x) {
        int p = atomicAdd(&pos[dom[i]], 1);
        perm[p] = i;
    }
}

// ------------- bf16 WMMA GEMM, 128x128 tile, async-to-LDS double buffered
// A: bf16 [rows x K] (row-gathered via perm), WT: bf16 [N x K] (+dom*wStride)
// 8 waves; wave = 32 rows x 64 cols = 8x v_wmma_f32_16x16x32_bf16 per K-step.
__global__ __launch_bounds__(256)
void k_gemm(const unsigned short* __restrict__ A, const unsigned short* __restrict__ WT,
            const float* __restrict__ bias,
            unsigned short* __restrict__ outB, float* __restrict__ outF,
            const int* __restrict__ perm, const int* __restrict__ tileDom,
            const unsigned short* __restrict__ zrow,
            int K, int N, int doRelu, long wStride, long bStride) {
    __shared__ __align__(16) unsigned short ldsA[2][BM * 48];   // 128 rows x 32 k (pad->48)
    __shared__ __align__(16) unsigned short ldsB[2][BM * 48];   // 128 cols x 32 k
    const int t    = threadIdx.x;
    const int lane = t & 31;
    const int wv   = t >> 5;              // 0..7
    const int rw   = wv & 3;              // row group (32 rows)
    const int cw   = wv >> 2;             // col group (64 cols)
    const int mrow = lane & 15;
    const int half = lane >> 4;
    const int m0   = blockIdx.x * BM;
    const int n0   = blockIdx.y * BM;
    const int dom  = tileDom ? tileDom[blockIdx.x] : 0;
    const unsigned short* Wd = WT + (size_t)dom * (size_t)wStride;

    // tile staging: each thread copies 2 A rows + 2 B rows, 16B each
    const int lr0 = t >> 2, lr1 = 64 + (t >> 2);
    const int colA = (t & 3) * 8;
    const int grow0 = perm ? perm[m0 + lr0] : (m0 + lr0);
    const int grow1 = perm ? perm[m0 + lr1] : (m0 + lr1);
    const unsigned short* pA0 = (grow0 < 0) ? (zrow + colA) : (A + (size_t)grow0 * K + colA);
    const unsigned short* pA1 = (grow1 < 0) ? (zrow + colA) : (A + (size_t)grow1 * K + colA);
    const unsigned short* pB0 = Wd + (size_t)(n0 + lr0) * K + colA;
    const unsigned short* pB1 = Wd + (size_t)(n0 + lr1) * K + colA;
    unsigned oA0[2], oA1[2], oB0[2], oB1[2];
#pragma unroll
    for (int b = 0; b < 2; ++b) {
        oA0[b] = (unsigned)(unsigned long long)&ldsA[b][lr0 * 48 + colA];
        oA1[b] = (unsigned)(unsigned long long)&ldsA[b][lr1 * 48 + colA];
        oB0[b] = (unsigned)(unsigned long long)&ldsB[b][lr0 * 48 + colA];
        oB1[b] = (unsigned)(unsigned long long)&ldsB[b][lr1 * 48 + colA];
    }
    auto issue = [&](int b, int k0) {
        asm volatile("global_load_async_to_lds_b128 %0, %1, off"
                     :: "v"(oA0[b]), "v"((unsigned long long)(pA0 + k0)) : "memory");
        asm volatile("global_load_async_to_lds_b128 %0, %1, off"
                     :: "v"(oA1[b]), "v"((unsigned long long)(pA1 + k0)) : "memory");
        asm volatile("global_load_async_to_lds_b128 %0, %1, off"
                     :: "v"(oB0[b]), "v"((unsigned long long)(pB0 + k0)) : "memory");
        asm volatile("global_load_async_to_lds_b128 %0, %1, off"
                     :: "v"(oB1[b]), "v"((unsigned long long)(pB1 + k0)) : "memory");
    };

    v8f acc[2][4] = {};
    const int nIter = K / 32;
    issue(0, 0);
    for (int i = 0; i < nIter; ++i) {
        const int buf = i & 1;
        if (i + 1 < nIter) {
            issue(buf ^ 1, (i + 1) * 32);
            asm volatile("s_wait_asynccnt 0x4" ::: "memory");   // current 4 done
        } else {
            asm volatile("s_wait_asynccnt 0x0" ::: "memory");
        }
        __syncthreads();
        // fragments (ISA 16-bit A layout: halves at k = h*8.. and 16+h*8..)
        v16bf af[2], bfv[4];
#pragma unroll
        for (int a = 0; a < 2; ++a) {
            int ar = rw * 32 + a * 16 + mrow;
            v8bf x0 = *(const v8bf*)&ldsA[buf][ar * 48 + half * 8];
            v8bf x1 = *(const v8bf*)&ldsA[buf][ar * 48 + 16 + half * 8];
#pragma unroll
            for (int q = 0; q < 8; ++q) { af[a][q] = x0[q]; af[a][q + 8] = x1[q]; }
        }
#pragma unroll
        for (int c = 0; c < 4; ++c) {
            int br = cw * 64 + c * 16 + mrow;
            v8bf x0 = *(const v8bf*)&ldsB[buf][br * 48 + half * 8];
            v8bf x1 = *(const v8bf*)&ldsB[buf][br * 48 + 16 + half * 8];
#pragma unroll
            for (int q = 0; q < 8; ++q) { bfv[c][q] = x0[q]; bfv[c][q + 8] = x1[q]; }
        }
#pragma unroll
        for (int a = 0; a < 2; ++a)
#pragma unroll
            for (int c = 0; c < 4; ++c)
                acc[a][c] = __builtin_amdgcn_wmma_f32_16x16x32_bf16(
                    false, af[a], false, bfv[c], (short)0, acc[a][c], false, false);
        __syncthreads();
    }
    // epilogue: C layout -> row = r + 8*half, col = mrow
#pragma unroll
    for (int a = 0; a < 2; ++a)
#pragma unroll
        for (int c = 0; c < 4; ++c) {
            int gn = n0 + cw * 64 + c * 16 + mrow;
            float bv = bias ? bias[(size_t)dom * bStride + gn] : 0.f;
#pragma unroll
            for (int r = 0; r < 8; ++r) {
                int orow = m0 + rw * 32 + a * 16 + r + 8 * half;
                float v = acc[a][c][r] + bv;
                if (doRelu) v = v > 0.f ? v : 0.f;
                size_t o = (size_t)orow * N + gn;
                if (outF) outF[o] = v;
                if (outB) outB[o] = f2bf(v);
            }
        }
}

// ---------------------------- row LayerNorm + scatter to up to 4 destinations
__global__ __launch_bounds__(128)
void k_ln(const float* __restrict__ y, const int* __restrict__ perm,
          const int* __restrict__ tileDom,
          const float* __restrict__ g, const float* __restrict__ b, long gbStride,
          float* __restrict__ o1, long rs1, float* __restrict__ o2, long rs2,
          unsigned short* __restrict__ ob1, long rb1,
          unsigned short* __restrict__ ob2, long rb2) {
    __shared__ float sm[128], sq[128];
    int row  = blockIdx.x;
    int orig = perm ? perm[row] : row;
    if (orig < 0) return;                       // padding row (uniform per block)
    int dom = tileDom ? tileDom[row / BM] : 0;
    int t = threadIdx.x;
    float x[6], s = 0.f, q = 0.f;
    const float* yr = y + (size_t)row * HH;
#pragma unroll
    for (int j = 0; j < 6; ++j) {
        x[j] = yr[t + j * 128];
        s += x[j]; q += x[j] * x[j];
    }
    sm[t] = s; sq[t] = q;
    __syncthreads();
    for (int off = 64; off > 0; off >>= 1) {
        if (t < off) { sm[t] += sm[t + off]; sq[t] += sq[t + off]; }
        __syncthreads();
    }
    float mean = sm[0] * (1.f / HH);
    float var  = sq[0] * (1.f / HH) - mean * mean;
    float rinv = rsqrtf(var + 1e-5f);
    const float* gd = g + (size_t)dom * gbStride;
    const float* bd = b + (size_t)dom * gbStride;
#pragma unroll
    for (int j = 0; j < 6; ++j) {
        int c = t + j * 128;
        float v = (x[j] - mean) * rinv * gd[c] + bd[c];
        if (o1)  o1[(size_t)orig * rs1 + c] = v;
        if (o2)  o2[(size_t)orig * rs2 + c] = v;
        if (ob1) ob1[(size_t)orig * rb1 + c] = f2bf(v);
        if (ob2) ob2[(size_t)orig * rb2 + c] = f2bf(v);
    }
}

// ------------------------------------------------ classifier tail (N=3, fp32)
__global__ void k_out3(const unsigned short* __restrict__ hc, const float* __restrict__ W,
                       const float* __restrict__ bias, float* __restrict__ out, int K) {
    int r = blockIdx.x * blockDim.x + threadIdx.x;
    if (r >= BB) return;
    float a0 = 0.f, a1 = 0.f, a2 = 0.f;
    const uint4* hr = (const uint4*)(hc + (size_t)r * K);
    for (int kk = 0; kk < K / 8; ++kk) {
        uint4 p = hr[kk];
        unsigned w[4] = {p.x, p.y, p.z, p.w};
#pragma unroll
        for (int qd = 0; qd < 4; ++qd) {
            int k = kk * 8 + qd * 2;
            float lo = bf2f_lo(w[qd]), hi = bf2f_hi(w[qd]);
            a0 += lo * W[k * 3 + 0] + hi * W[(k + 1) * 3 + 0];
            a1 += lo * W[k * 3 + 1] + hi * W[(k + 1) * 3 + 1];
            a2 += lo * W[k * 3 + 2] + hi * W[(k + 1) * 3 + 2];
        }
    }
    out[(size_t)r * 3 + 0] = a0 + bias[0];
    out[(size_t)r * 3 + 1] = a1 + bias[1];
    out[(size_t)r * 3 + 2] = a2 + bias[2];
}

// ------------------------- discriminator tail (N=5) + log_softmax + NLL loss
__global__ void k_disc(const unsigned short* __restrict__ hd, const float* __restrict__ W,
                       const float* __restrict__ bias, const int* __restrict__ dom,
                       float* __restrict__ dp, float* __restrict__ lossAcc, int K) {
    int r = blockIdx.x * blockDim.x + threadIdx.x;
    if (r >= BB) return;
    float a[DD] = {};
    const uint4* hr = (const uint4*)(hd + (size_t)r * K);
    for (int kk = 0; kk < K / 8; ++kk) {
        uint4 p = hr[kk];
        unsigned w[4] = {p.x, p.y, p.z, p.w};
#pragma unroll
        for (int qd = 0; qd < 4; ++qd) {
            int k = kk * 8 + qd * 2;
            float lo = bf2f_lo(w[qd]), hi = bf2f_hi(w[qd]);
#pragma unroll
            for (int j = 0; j < DD; ++j)
                a[j] += lo * W[k * DD + j] + hi * W[(k + 1) * DD + j];
        }
    }
    float m = -1e30f;
#pragma unroll
    for (int j = 0; j < DD; ++j) {
        a[j] += bias[j];
        dp[(size_t)r * DD + j] = a[j];
        m = fmaxf(m, a[j]);
    }
    float sum = 0.f;
#pragma unroll
    for (int j = 0; j < DD; ++j) sum += __expf(a[j] - m);
    float lse = m + __logf(sum);
    atomicAdd(lossAcc, (lse - a[dom[r]]) * (1.0f / BB));
}

__global__ void k_loss(const float* __restrict__ acc, float* __restrict__ out) {
    out[0] = acc[0];
}

// ------------------------------------------------------------------ launcher
extern "C" void kernel_launch(void* const* d_in, const int* in_sizes, int n_in,
                              void* d_out, int out_size, void* d_ws, size_t ws_size,
                              hipStream_t stream) {
    (void)in_sizes; (void)n_in; (void)out_size; (void)ws_size;
    const float* features = (const float*)d_in[0];
    const int*   domIds   = (const int*)d_in[1];
    const float* dW1  = (const float*)d_in[2],  *dB1  = (const float*)d_in[3];
    const float* dW2  = (const float*)d_in[4],  *dB2  = (const float*)d_in[5];
    const float* dLNg = (const float*)d_in[6],  *dLNb = (const float*)d_in[7];
    const float* iW1  = (const float*)d_in[8],  *iB1  = (const float*)d_in[9];
    const float* iW2  = (const float*)d_in[10], *iB2  = (const float*)d_in[11];
    const float* iLNg = (const float*)d_in[12], *iLNb = (const float*)d_in[13];
    const float* discW1 = (const float*)d_in[14], *discB1 = (const float*)d_in[15];
    const float* discW2 = (const float*)d_in[16], *discB2 = (const float*)d_in[17];
    const float* clsW1  = (const float*)d_in[18], *clsB1  = (const float*)d_in[19];
    const float* clsW2  = (const float*)d_in[20], *clsB2  = (const float*)d_in[21];

    char* ws = (char*)d_ws;
    size_t off = 0;
    auto alloc = [&](size_t bytes) {
        void* p = ws + off;
        off = (off + bytes + 255) & ~(size_t)255;
        return p;
    };
    unsigned short* featB  = (unsigned short*)alloc((size_t)BB * HH * 2);
    unsigned short* hbuf   = (unsigned short*)alloc((size_t)MPAD * HH * 2);
    float*          ybuf   = (float*)alloc((size_t)MPAD * HH * 4);
    unsigned short* invB   = (unsigned short*)alloc((size_t)BB * HH * 2);
    unsigned short* combB  = (unsigned short*)alloc((size_t)BB * 2 * HH * 2);
    unsigned short* hdB    = (unsigned short*)alloc((size_t)BB * 384 * 2);
    unsigned short* iW1T   = (unsigned short*)alloc((size_t)HH * HH * 2);
    unsigned short* iW2T   = (unsigned short*)alloc((size_t)HH * HH * 2);
    unsigned short* dW1T   = (unsigned short*)alloc((size_t)DD * HH * HH * 2);
    unsigned short* dW2T   = (unsigned short*)alloc((size_t)DD * HH * HH * 2);
    unsigned short* clsW1T = (unsigned short*)alloc((size_t)2 * HH * HH * 2);
    unsigned short* discW1T= (unsigned short*)alloc((size_t)HH * 384 * 2);
    unsigned short* zrow   = (unsigned short*)alloc((size_t)2 * HH * 2);
    int*   perm    = (int*)alloc((size_t)MPAD * 4);
    int*   tileD   = (int*)alloc((size_t)MT * 4);
    float* lossAcc = (float*)alloc(256);

    float* out     = (float*)d_out;
    float* oAspect = out;
    float* oDfeat  = oAspect + (size_t)BB * 3;
    float* oInv    = oDfeat + (size_t)BB * HH;
    float* oComb   = oInv + (size_t)BB * HH;
    float* oLoss   = oComb + (size_t)BB * 2 * HH;
    float* oDpred  = oLoss + 1;

    k_cvt<<<1024, 256, 0, stream>>>(features, featB, BB * HH);
    k_sort<<<1, 1024, 0, stream>>>(domIds, perm, tileD, lossAcc, zrow);
    // one-time weight transpose+bf16 (W^T [N,K])
    dim3 tb(32, 8);
    k_cvtT<<<dim3(HH / 32, HH / 32, 1),  tb, 0, stream>>>(iW1, iW1T, HH, HH);
    k_cvtT<<<dim3(HH / 32, HH / 32, 1),  tb, 0, stream>>>(iW2, iW2T, HH, HH);
    k_cvtT<<<dim3(HH / 32, HH / 32, DD), tb, 0, stream>>>(dW1, dW1T, HH, HH);
    k_cvtT<<<dim3(HH / 32, HH / 32, DD), tb, 0, stream>>>(dW2, dW2T, HH, HH);
    k_cvtT<<<dim3(HH / 32, 2 * HH / 32, 1), tb, 0, stream>>>(clsW1, clsW1T, 2 * HH, HH);
    k_cvtT<<<dim3(384 / 32, HH / 32, 1), tb, 0, stream>>>(discW1, discW1T, HH, 384);

    dim3 blk(256);
    // invariant encoder: relu(x@iW1+b) @ iW2 + b -> LN
    k_gemm<<<dim3(BB / BM, HH / BM), blk, 0, stream>>>(featB, iW1T, iB1, hbuf, nullptr,
        nullptr, nullptr, zrow, HH, HH, 1, 0, 0);
    k_gemm<<<dim3(BB / BM, HH / BM), blk, 0, stream>>>(hbuf, iW2T, iB2, nullptr, ybuf,
        nullptr, nullptr, zrow, HH, HH, 0, 0, 0);
    k_ln<<<BB, 128, 0, stream>>>(ybuf, nullptr, nullptr, iLNg, iLNb, 0,
        oInv, HH, oComb + HH, 2 * HH, invB, HH, combB + HH, 2 * HH);
    // domain experts: grouped (row-gathered) GEMMs, only each row's own expert
    k_gemm<<<dim3(MT, HH / BM), blk, 0, stream>>>(featB, dW1T, dB1, hbuf, nullptr,
        perm, tileD, zrow, HH, HH, 1, (long)HH * HH, HH);
    k_gemm<<<dim3(MT, HH / BM), blk, 0, stream>>>(hbuf, dW2T, dB2, nullptr, ybuf,
        nullptr, tileD, zrow, HH, HH, 0, (long)HH * HH, HH);
    k_ln<<<MPAD, 128, 0, stream>>>(ybuf, perm, tileD, dLNg, dLNb, HH,
        oDfeat, HH, oComb, 2 * HH, combB, 2 * HH, nullptr, 0);
    // aspect classifier
    k_gemm<<<dim3(BB / BM, HH / BM), blk, 0, stream>>>(combB, clsW1T, clsB1, hbuf, nullptr,
        nullptr, nullptr, zrow, 2 * HH, HH, 1, 0, 0);
    k_out3<<<BB / 256, 256, 0, stream>>>(hbuf, clsW2, clsB2, oAspect, HH);
    // domain discriminator
    k_gemm<<<dim3(BB / BM, 384 / BM), blk, 0, stream>>>(invB, discW1T, discB1, hdB, nullptr,
        nullptr, nullptr, zrow, HH, 384, 1, 0, 0);
    k_disc<<<BB / 256, 256, 0, stream>>>(hdB, discW2, discB2, domIds, oDpred, lossAcc, 384);
    k_loss<<<1, 1, 0, stream>>>(lossAcc, oLoss);
}
